// AUGRUCell_38276748542530
// MI455X (gfx1250) — matrix operand order repaired and checked by reading, not compiled, not run on hardware
//
#include <hip/hip_runtime.h>

typedef __attribute__((ext_vector_type(16))) _Float16 v16h;
typedef __attribute__((ext_vector_type(8)))  _Float16 v8h;
typedef __attribute__((ext_vector_type(4)))  _Float16 v4h;
typedef __attribute__((ext_vector_type(8)))  float    v8f;
typedef __attribute__((ext_vector_type(4)))  float    v4f;

#define HID        128
#define NT_TOTAL   24      // 384 gate columns / 16
#define FRAG_HALFS 512     // 32 lanes * 16 halfs per WMMA B fragment
#define SX_STRIDE  136     // 128 + 8 pad (f16 rows)
#define SF_STRIDE  132     // 128 + 4 pad (f32 rows)

__device__ __forceinline__ v8f wmma_f16(v16h a, v16h b, v8f c) {
  // v_wmma_f32_16x16x32_f16: D = A(16x32 f16) * B(32x16 f16) + C(16x16 f32)
  return __builtin_amdgcn_wmma_f32_16x16x32_f16(false, a, false, b, (short)0, c,
                                                false, false);
}

__device__ __forceinline__ float sigmoid_fast(float t) {
  return 1.0f / (1.0f + __expf(-t));
}
__device__ __forceinline__ float tanh_fast(float t) {
  float e = __expf(-2.0f * t);
  return (1.0f - e) / (1.0f + e);
}

// ---------------------------------------------------------------------------
// Prep: convert f32 weights into f16 B-fragment blobs, pre-swizzled to the
// WMMA B-matrix register layout so the main kernel loads them with two
// contiguous global_load_b128 per lane.
// B layout (32x16 K x N, 16-bit): lanes 0-15 hold K=0..7,16..23 (N = lane),
// lanes 16-31 hold K=8..15,24..31 (N = lane-16), 2 halfs per VGPR in K order.
// ---------------------------------------------------------------------------
__global__ __launch_bounds__(256) void augru_prep(const float* __restrict__ Wx,
                                                  const float* __restrict__ Wh,
                                                  _Float16* __restrict__ blob) {
  int t = blockIdx.x * 256 + threadIdx.x;
  if (t >= 2 * NT_TOTAL * 4 * FRAG_HALFS) return;
  int e = t & 15;
  int r = t >> 4;
  int lane = r & 31; r >>= 5;
  int ks   = r & 3;  r >>= 2;
  int nt   = r % NT_TOTAL;
  int w    = r / NT_TOTAL;
  const float* W = w ? Wh : Wx;                 // row-major (384, 128)
  int g = nt * 16 + (lane & 15);                // gate column (N)
  int i = ks * 32 + (e & 8) * 2 + (lane >> 4) * 8 + (e & 7);  // input dim (K)
  blob[t] = (_Float16)W[g * HID + i];
}

__device__ __forceinline__ v16h load_bfrag(const _Float16* __restrict__ blob,
                                           int w, int nt, int ks, int lane) {
  int fi = (w * NT_TOTAL + nt) * 4 + ks;
  return *(const v16h*)(blob + fi * FRAG_HALFS + lane * 16);
}

// ---------------------------------------------------------------------------
// Main fused AUGRU kernel. 128 threads (4 waves), 64 batch rows per block.
// Each wave owns a 16-row M tile and computes all 384 gate columns via WMMA,
// then blends u/r/h_tilde + attention entirely in registers.
// ---------------------------------------------------------------------------
__global__ __launch_bounds__(128) void augru_main(
    const float* __restrict__ x, const float* __restrict__ hprev,
    const float* __restrict__ att,
    const float* __restrict__ bx, const float* __restrict__ bh,
    const _Float16* __restrict__ blob, float* __restrict__ out) {
  __shared__ _Float16 sx[64 * SX_STRIDE];   // x tile, f16
  __shared__ _Float16 sh[64 * SX_STRIDE];   // h tile, f16 (A fragments)
  __shared__ float    shf[64 * SF_STRIDE];  // h tile, f32 (final blend)

  const int tid = threadIdx.x;
  const long rowbase = (long)blockIdx.x * 64;

  // ---- cooperative staging: coalesced b128 streaming loads, f32->f16 ----
#pragma unroll
  for (int it = 0; it < 16; ++it) {
    int c   = it * 128 + tid;   // float4 chunk id: 64 rows * 32 chunks
    int row = c >> 5;
    int c4  = c & 31;
    const float* xp = x     + (rowbase + row) * HID + c4 * 4;
    const float* hp = hprev + (rowbase + row) * HID + c4 * 4;
    v4f xv = __builtin_nontemporal_load((const v4f*)xp);
    v4f hv = __builtin_nontemporal_load((const v4f*)hp);
    v4h x4 = {(_Float16)xv.x, (_Float16)xv.y, (_Float16)xv.z, (_Float16)xv.w};
    v4h h4 = {(_Float16)hv.x, (_Float16)hv.y, (_Float16)hv.z, (_Float16)hv.w};
    *(v4h*)(sx + row * SX_STRIDE + c4 * 4)  = x4;
    *(v4h*)(sh + row * SX_STRIDE + c4 * 4)  = h4;
    *(v4f*)(shf + row * SF_STRIDE + c4 * 4) = hv;
  }
  __syncthreads();

  const int lane = tid & 31;
  const int wave = tid >> 5;
  const int half = lane >> 4;
  const int n    = lane & 15;           // N column within tile / A-matrix M row
  const int lrow = wave * 16 + n;       // LDS row this lane gathers (A matrix)

  // ---- build A fragments (ISA 16-bit A 16x32 layout) ----
  // lanes 0-15:  e0..7 -> K = k1..k1+7 (k1 = 0),  e8..15 -> K = k1+16..
  // lanes 16-31: runs shifted by 8 (k1 = 8)
  v16h ax[4], ah[4];
#pragma unroll
  for (int ks = 0; ks < 4; ++ks) {
    int k1 = ks * 32 + 8 * half;
    v8h xlo = *(const v8h*)(sx + lrow * SX_STRIDE + k1);
    v8h xhi = *(const v8h*)(sx + lrow * SX_STRIDE + k1 + 16);
    v8h hlo = *(const v8h*)(sh + lrow * SX_STRIDE + k1);
    v8h hhi = *(const v8h*)(sh + lrow * SX_STRIDE + k1 + 16);
    v16h a, b;
#pragma unroll
    for (int i = 0; i < 8; ++i) {
      a[i] = xlo[i]; a[i + 8] = xhi[i];
      b[i] = hlo[i]; b[i + 8] = hhi[i];
    }
    ax[ks] = a; ah[ks] = b;
  }

  // ---- phase 1: r gate (tiles 8..15), sigmoid in-register ----
  v8f rg[8];
#pragma unroll
  for (int j = 0; j < 8; ++j) {
    int nt = 8 + j;
    v8f acc = {};
#pragma unroll
    for (int ks = 0; ks < 4; ++ks) {
      acc = wmma_f16(ax[ks], load_bfrag(blob, 0, nt, ks, lane), acc);
      acc = wmma_f16(ah[ks], load_bfrag(blob, 1, nt, ks, lane), acc);
    }
    float bias = bx[nt * 16 + n] + bh[nt * 16 + n];
#pragma unroll
    for (int v = 0; v < 8; ++v) rg[j][v] = sigmoid_fast(acc[v] + bias);
  }

  // ---- phase 2: candidate h_tilde = tanh(x_h + b_xh + r*(h_h + b_hh)) ----
  v8f ht[8];
#pragma unroll
  for (int j = 0; j < 8; ++j) {
    int nt = 16 + j;
    v8f xa = {}, ha = {};
#pragma unroll
    for (int ks = 0; ks < 4; ++ks) {
      xa = wmma_f16(ax[ks], load_bfrag(blob, 0, nt, ks, lane), xa);
      ha = wmma_f16(ah[ks], load_bfrag(blob, 1, nt, ks, lane), ha);
    }
    float bxi = bx[nt * 16 + n];
    float bhi = bh[nt * 16 + n];
#pragma unroll
    for (int v = 0; v < 8; ++v)
      ht[j][v] = tanh_fast(xa[v] + bxi + rg[j][v] * (ha[v] + bhi));
  }

  // attention scores for this wave's 16 rows (broadcast loads)
  float av[8];
#pragma unroll
  for (int v = 0; v < 8; ++v)
    av[v] = att[rowbase + wave * 16 + half * 8 + v];

  // ---- phase 3: u gate (tiles 0..7) + attention blend + store ----
#pragma unroll
  for (int j = 0; j < 8; ++j) {
    int nt = j;
    v8f acc = {};
#pragma unroll
    for (int ks = 0; ks < 4; ++ks) {
      acc = wmma_f16(ax[ks], load_bfrag(blob, 0, nt, ks, lane), acc);
      acc = wmma_f16(ah[ks], load_bfrag(blob, 1, nt, ks, lane), acc);
    }
    float bias = bx[nt * 16 + n] + bh[nt * 16 + n];
#pragma unroll
    for (int v = 0; v < 8; ++v) {
      int mrow = wave * 16 + half * 8 + v;      // C/D layout: M = v + 8*half
      float u  = sigmoid_fast(acc[v] + bias);
      float ua = av[v] * u;
      float hp = shf[mrow * SF_STRIDE + j * 16 + n];  // f32 h_prev from LDS
      float hn = (1.0f - ua) * hp + ua * ht[j][v];
      float* op = out + (rowbase + mrow) * HID + j * 16 + n;
      __builtin_nontemporal_store(hn, op);
    }
  }
}

extern "C" void kernel_launch(void* const* d_in, const int* in_sizes, int n_in,
                              void* d_out, int out_size, void* d_ws, size_t ws_size,
                              hipStream_t stream) {
  const float* x   = (const float*)d_in[0];
  const float* h   = (const float*)d_in[1];
  const float* att = (const float*)d_in[2];
  const float* Wx  = (const float*)d_in[3];
  const float* bx  = (const float*)d_in[4];
  const float* Wh  = (const float*)d_in[5];
  const float* bh  = (const float*)d_in[6];
  float* out = (float*)d_out;
  _Float16* blob = (_Float16*)d_ws;   // 192 KB of pre-swizzled f16 B fragments

  const int B = in_sizes[2];          // att_score length == batch
  const int nfrag_elems = 2 * NT_TOTAL * 4 * FRAG_HALFS;  // 98304

  augru_prep<<<(nfrag_elems + 255) / 256, 256, 0, stream>>>(Wx, Wh, blob);
  augru_main<<<B / 64, 128, 0, stream>>>(x, h, att, bx, bh, blob, out);
}
